// Attention_3582002725378
// MI455X (gfx1250) — matrix operand order repaired and checked by reading, not compiled, not run on hardware
//
#include <hip/hip_runtime.h>
#include <hip/hip_bf16.h>

// MHA forward for B=2,N=2048,DIM=256,H=4,DH=32 on gfx1250 (MI455X).
// bf16 WMMA (v_wmma_f32_16x16x32_bf16), f32 accumulate, flash-attention
// online softmax. K/V tiles double-buffered in LDS via
// GLOBAL_LOAD_ASYNC_TO_LDS_B128 + s_wait_asynccnt. pos_bias (67MB, the
// bandwidth-dominant stream) read with non-temporal hint.

#define BB   2
#define NN   2048
#define DIMM 256
#define HH   4
#define DHH  32

typedef __bf16 bf16_t;
typedef __attribute__((ext_vector_type(16))) __bf16 v16bf;
typedef __attribute__((ext_vector_type(8)))  __bf16 v8bf;
typedef __attribute__((ext_vector_type(8)))  float  v8f;

// Build a 16-element bf16 A/B fragment from two 16-byte contiguous runs.
__device__ __forceinline__ v16bf frag16(const bf16_t* lo, const bf16_t* hi) {
    v8bf a = *(const v8bf*)lo;
    v8bf b = *(const v8bf*)hi;
    v16bf r;
#pragma unroll
    for (int i = 0; i < 8; ++i) { r[i] = a[i]; r[i + 8] = b[i]; }
    return r;
}

__device__ __forceinline__ float rowmax16(float x) {
#pragma unroll
    for (int m = 1; m < 16; m <<= 1) x = fmaxf(x, __shfl_xor(x, m, 32));
    return x;
}
__device__ __forceinline__ float rowsum16(float x) {
#pragma unroll
    for (int m = 1; m < 16; m <<= 1) x += __shfl_xor(x, m, 32);
    return x;
}

// ---- CDNA5 async global->LDS (ASYNCcnt) -------------------------------
__device__ __forceinline__ void async_b128_to_lds(unsigned int lds_off,
                                                  const void* gptr) {
    unsigned long long g = (unsigned long long)(uintptr_t)gptr;
    asm volatile("global_load_async_to_lds_b128 %0, %1, off"
                 :: "v"(lds_off), "v"(g) : "memory");
}
__device__ __forceinline__ void wait_async0() {
#if __has_builtin(__builtin_amdgcn_s_wait_asynccnt)
    __builtin_amdgcn_s_wait_asynccnt(0);
#else
    asm volatile("s_wait_asynccnt 0x0" ::: "memory");
#endif
}

// ---------------- cast kernels ----------------
__global__ void __launch_bounds__(256)
cvt_f32_bf16(const float* __restrict__ src, bf16_t* __restrict__ dst, int n) {
    int i = blockIdx.x * 256 + threadIdx.x;
    if (i < n) dst[i] = (bf16_t)src[i];
}

// w[K][N] (f32) -> wt[N][K] (bf16): makes WMMA B-fragments contiguous.
__global__ void __launch_bounds__(256)
cvt_transpose_bf16(const float* __restrict__ w, bf16_t* __restrict__ wt, int K, int N) {
    int i = blockIdx.x * 256 + threadIdx.x;
    if (i < K * N) {
        int k = i / N, n = i % N;
        wt[n * K + k] = (bf16_t)w[i];
    }
}

// ---------------- QKV projection ----------------
// qkv = Xb[4096,256] @ W; one 16x16 tile per wave. Q scaled by DH^-0.5.
// Q,K stored [B,H,N,DH]; V stored TRANSPOSED [B,H,DH,N] so attention can
// async-copy V^T tiles straight into LDS with no transpose pass.
__global__ void __launch_bounds__(256)
qkv_gemm(const bf16_t* __restrict__ Xb, const bf16_t* __restrict__ WtQ,
         bf16_t* __restrict__ Q, bf16_t* __restrict__ K, bf16_t* __restrict__ Vt) {
    const int lane = threadIdx.x & 31, wave = threadIdx.x >> 5;
    const int ln = lane & 15, lh = lane >> 4;
    const int rowbase = (blockIdx.y * 8 + wave) * 16;   // M = B*N = 4096
    const int colbase = blockIdx.x * 16;                // Nout = 384

    const bf16_t* Arow = Xb  + (size_t)(rowbase + ln) * DIMM;
    const bf16_t* Brow = WtQ + (size_t)(colbase + ln) * DIMM;

    v8f c;
#pragma unroll
    for (int i = 0; i < 8; ++i) c[i] = 0.f;

#pragma unroll
    for (int it = 0; it < DIMM / 32; ++it) {
        v16bf a = frag16(Arow + it * 32 + 8 * lh, Arow + it * 32 + 16 + 8 * lh);
        v16bf b = frag16(Brow + it * 32 + 8 * lh, Brow + it * 32 + 16 + 8 * lh);
        c = __builtin_amdgcn_wmma_f32_16x16x32_bf16(false, a, false, b,
                                                    (short)0, c, false, false);
    }

    const float scale = 0.17677669529663687f; // DH^-0.5
#pragma unroll
    for (int v = 0; v < 8; ++v) {
        int row  = rowbase + v + 8 * lh;
        int col  = colbase + ln;
        int part = col >> 7;            // 0=q 1=k 2=v
        int head = (col >> 5) & 3;
        int d    = col & 31;
        int bi   = row >> 11;
        int nr   = row & (NN - 1);
        float val = c[v];
        if (part == 0) {
            Q[(size_t)((bi * HH + head) * NN + nr) * DHH + d] = (bf16_t)(val * scale);
        } else if (part == 1) {
            K[(size_t)((bi * HH + head) * NN + nr) * DHH + d] = (bf16_t)val;
        } else {
            Vt[(size_t)((bi * HH + head) * DHH + d) * NN + nr] = (bf16_t)val;
        }
    }
}

// ---------------- flash attention ----------------
#define KSTR 40   // padded LDS row stride (elements): 80B rows, 16B aligned
__global__ void __launch_bounds__(128)
attn(const bf16_t* __restrict__ Q, const bf16_t* __restrict__ K,
     const bf16_t* __restrict__ Vt, const float* __restrict__ pb,
     const unsigned char* __restrict__ fpm, bf16_t* __restrict__ O) {
    __shared__ __align__(16) bf16_t Ksh [2][32 * KSTR];  // [key][d]
    __shared__ __align__(16) bf16_t Vtsh[2][32 * KSTR];  // [d][key]
    __shared__ __align__(16) bf16_t Psh [4][16 * KSTR];  // per-wave P tiles

    const int lane = threadIdx.x & 31, wave = threadIdx.x >> 5;
    const int ln = lane & 15, lh = lane >> 4;
    const int b = blockIdx.z, h = blockIdx.y;
    const int qbase = blockIdx.x * 64 + wave * 16;
    const bool fm = fpm[b] != 0;

    const bf16_t* Kb  = K  + (size_t)(b * HH + h) * NN * DHH;   // [key][d]
    const bf16_t* Vtb = Vt + (size_t)(b * HH + h) * DHH * NN;   // [d][key]
    const float*  pbh = pb + (size_t)h * NN * NN;

    // Q A-fragment: rows qbase+ln, d runs [8lh,8lh+8) and [16+8lh,...)
    const bf16_t* Qrow = Q + (size_t)((b * HH + h) * NN + qbase + ln) * DHH;
    v16bf qa = frag16(Qrow + 8 * lh, Qrow + 16 + 8 * lh);

    v8f o0, o1;
    float rmax[8], rsum[8];
#pragma unroll
    for (int v = 0; v < 8; ++v) { o0[v] = 0.f; o1[v] = 0.f; rmax[v] = -3.0e38f; rsum[v] = 0.f; }

    // async stage of one 32-key tile into buffer `buf` (one B128/lane/operand)
    auto stage = [&](int buf, int jt) {
        const int t = threadIdx.x;
        {   // K tile: 32 keys x 32 d, rows contiguous in global
            int key = t >> 2, dc = (t & 3) * 8;
            async_b128_to_lds((unsigned int)(uintptr_t)&Ksh[buf][key * KSTR + dc],
                              Kb + (size_t)(jt * 32 + key) * DHH + dc);
        }
        {   // V^T tile: 32 d x 32 keys, rows stride NN in global
            int d = t >> 2, kc = (t & 3) * 8;
            async_b128_to_lds((unsigned int)(uintptr_t)&Vtsh[buf][d * KSTR + kc],
                              Vtb + (size_t)d * NN + jt * 32 + kc);
        }
    };

    stage(0, 0);
    for (int jt = 0; jt < NN / 32; ++jt) {
        const int cur = jt & 1;
        wait_async0();          // this wave's async copies into `cur` complete
        __syncthreads();        // everyone's copies visible; prev reads done
        if (jt + 1 < NN / 32) stage(cur ^ 1, jt + 1);   // prefetch next tile

        const bf16_t* Kc = Ksh[cur];
        const bf16_t* Vc = Vtsh[cur];

        // S = Q(16x32) x K^T(32x32) -> two 16x16 accumulators
        v8f s0, s1;
#pragma unroll
        for (int i = 0; i < 8; ++i) { s0[i] = 0.f; s1[i] = 0.f; }
        {
            v16bf kb0 = frag16(&Kc[ln * KSTR + 8 * lh],        &Kc[ln * KSTR + 16 + 8 * lh]);
            v16bf kb1 = frag16(&Kc[(16 + ln) * KSTR + 8 * lh], &Kc[(16 + ln) * KSTR + 16 + 8 * lh]);
            s0 = __builtin_amdgcn_wmma_f32_16x16x32_bf16(false, qa, false, kb0, (short)0, s0, false, false);
            s1 = __builtin_amdgcn_wmma_f32_16x16x32_bf16(false, qa, false, kb1, (short)0, s1, false, false);
        }

        // bias + mask + online softmax; write P (bf16) to wave-private LDS
#pragma unroll
        for (int v = 0; v < 8; ++v) {
            int rq = qbase + v + 8 * lh;
            int c0 = jt * 32 + ln;
            int c1 = c0 + 16;
            float sv0 = s0[v] + __builtin_nontemporal_load(&pbh[(size_t)rq * NN + c0]);
            float sv1 = s1[v] + __builtin_nontemporal_load(&pbh[(size_t)rq * NN + c1]);
            if (fm) {                      // attend only to self
                if (rq != c0) sv0 = -1.0e30f;
                if (rq != c1) sv1 = -1.0e30f;
            }
            float cm = rowmax16(fmaxf(sv0, sv1));
            float mn = fmaxf(rmax[v], cm);
            float alpha = __expf(rmax[v] - mn);
            rmax[v] = mn;
            float e0 = __expf(sv0 - mn);
            float e1 = __expf(sv1 - mn);
            rsum[v] = rsum[v] * alpha + rowsum16(e0 + e1);
            o0[v] *= alpha; o1[v] *= alpha;
            Psh[wave][(v + 8 * lh) * KSTR + ln]      = (bf16_t)e0;
            Psh[wave][(v + 8 * lh) * KSTR + 16 + ln] = (bf16_t)e1;
        }

        // O += P(16x32) x V(32x32): P re-read in A layout (same-wave LDS, in order)
        const bf16_t* Prow = &Psh[wave][ln * KSTR];
        v16bf pa  = frag16(Prow + 8 * lh, Prow + 16 + 8 * lh);
        v16bf vb0 = frag16(&Vc[ln * KSTR + 8 * lh],        &Vc[ln * KSTR + 16 + 8 * lh]);
        v16bf vb1 = frag16(&Vc[(16 + ln) * KSTR + 8 * lh], &Vc[(16 + ln) * KSTR + 16 + 8 * lh]);
        o0 = __builtin_amdgcn_wmma_f32_16x16x32_bf16(false, pa, false, vb0, (short)0, o0, false, false);
        o1 = __builtin_amdgcn_wmma_f32_16x16x32_bf16(false, pa, false, vb1, (short)0, o1, false, false);
        // no trailing barrier needed: next iteration's wait+barrier orders
        // this buffer's reads against its async overwrite.
    }

    // normalize + store merged heads [B*N, H*DH] bf16
#pragma unroll
    for (int v = 0; v < 8; ++v) {
        float inv = 1.0f / rsum[v];
        int row = qbase + v + 8 * lh;
        size_t base = (size_t)(b * NN + row) * (HH * DHH) + h * DHH;
        O[base + ln]      = (bf16_t)(o0[v] * inv);
        O[base + 16 + ln] = (bf16_t)(o1[v] * inv);
    }
}

// ---------------- output projection ----------------
__global__ void __launch_bounds__(256)
out_gemm(const bf16_t* __restrict__ Ob, const bf16_t* __restrict__ WtO,
         float* __restrict__ out) {
    const int lane = threadIdx.x & 31, wave = threadIdx.x >> 5;
    const int ln = lane & 15, lh = lane >> 4;
    const int rowbase = (blockIdx.y * 8 + wave) * 16;  // M = 4096
    const int colbase = blockIdx.x * 16;               // N = 256

    const bf16_t* Arow = Ob  + (size_t)(rowbase + ln) * (HH * DHH);
    const bf16_t* Brow = WtO + (size_t)(colbase + ln) * (HH * DHH);

    v8f c;
#pragma unroll
    for (int i = 0; i < 8; ++i) c[i] = 0.f;

#pragma unroll
    for (int it = 0; it < (HH * DHH) / 32; ++it) {
        v16bf a = frag16(Arow + it * 32 + 8 * lh, Arow + it * 32 + 16 + 8 * lh);
        v16bf b = frag16(Brow + it * 32 + 8 * lh, Brow + it * 32 + 16 + 8 * lh);
        c = __builtin_amdgcn_wmma_f32_16x16x32_bf16(false, a, false, b,
                                                    (short)0, c, false, false);
    }
#pragma unroll
    for (int v = 0; v < 8; ++v)
        out[(size_t)(rowbase + v + 8 * lh) * DIMM + colbase + ln] = c[v];
}

// ---------------- host launcher ----------------
extern "C" void kernel_launch(void* const* d_in, const int* in_sizes, int n_in,
                              void* d_out, int out_size, void* d_ws, size_t ws_size,
                              hipStream_t stream) {
    (void)in_sizes; (void)n_in; (void)out_size; (void)ws_size;
    const float* x    = (const float*)d_in[0];
    const float* pb   = (const float*)d_in[1];
    const unsigned char* fm = (const unsigned char*)d_in[2];
    const float* wqkv = (const float*)d_in[3];
    const float* wout = (const float*)d_in[4];

    char* ws = (char*)d_ws;
    bf16_t* Xb  = (bf16_t*)(ws + 0);          // 2,097,152 B
    bf16_t* WtQ = (bf16_t*)(ws + 2097152);    //   196,608 B ([384][256])
    bf16_t* WtO = (bf16_t*)(ws + 2293760);    //    65,536 B ([256][128])
    bf16_t* Qb  = (bf16_t*)(ws + 2359296);    // 1,048,576 B [B,H,N,DH]
    bf16_t* Kb  = (bf16_t*)(ws + 3407872);    // 1,048,576 B [B,H,N,DH]
    bf16_t* Vtb = (bf16_t*)(ws + 4456448);    // 1,048,576 B [B,H,DH,N] (transposed)
    bf16_t* Ob  = (bf16_t*)(ws + 5505024);    // 1,048,576 B [B*N, H*DH]

    const int nx = BB * NN * DIMM;            // 1,048,576
    cvt_f32_bf16<<<(nx + 255) / 256, 256, 0, stream>>>(x, Xb, nx);
    cvt_transpose_bf16<<<(DIMM * 3 * HH * DHH + 255) / 256, 256, 0, stream>>>(
        wqkv, WtQ, DIMM, 3 * HH * DHH);
    cvt_transpose_bf16<<<(HH * DHH * DIMM + 255) / 256, 256, 0, stream>>>(
        wout, WtO, HH * DHH, DIMM);

    qkv_gemm<<<dim3(24, 32), 256, 0, stream>>>(Xb, WtQ, Qb, Kb, Vtb);
    attn<<<dim3(NN / 64, HH, BB), 128, 0, stream>>>(Qb, Kb, Vtb, pb, fm, Ob);
    out_gemm<<<dim3(16, 32), 256, 0, stream>>>(Ob, WtO, (float*)d_out);
}
